// Attention2D_35605278883930
// MI455X (gfx1250) — matrix operand order repaired and checked by reading, not compiled, not run on hardware
//
#include <hip/hip_runtime.h>
#include <hip/hip_bf16.h>

typedef __attribute__((ext_vector_type(16))) __bf16 v16bf;
typedef __attribute__((ext_vector_type(8)))  __bf16 v8bf;
typedef __attribute__((ext_vector_type(8)))  float  v8f;

#define WMMA_BF16(a, b, c) \
  __builtin_amdgcn_wmma_f32_16x16x32_bf16(false, (a), false, (b), (short)0, (c), false, false)

// ---------------------------------------------------------------------------
// A-fragment (16xK tile, 16-bit): lane L<16 holds row M=L, K = {0..7, 16..23};
// lane L+16 holds row M=L, K = {8..15, 24..31}  (CDNA5 ISA 7.12.2)
// ---------------------------------------------------------------------------
__device__ static inline v16bf load_a16(const __bf16* base, int pitch, int lane, int k0) {
    int r = lane & 15, half = lane >> 4;
    const __bf16* p = base + r * pitch + k0 + half * 8;
    union { v16bf v; v8bf h[2]; } u;
    u.h[0] = *(const v8bf*)(p);        // K = k0 + half*8 + 0..7
    u.h[1] = *(const v8bf*)(p + 16);   // K = k0 + 16 + half*8 + 0..7
    return u.v;
}

// ---------------------------------------------------------------------------
// B-fragment (Kx16 tile, 16-bit): lane L holds column N = n0 + (L&15),
// K = k0 + (L>>4)*8 + {0..7, 16..23} from WT[n][k] (pre-transposed)
// ---------------------------------------------------------------------------
__device__ static inline v16bf load_b16(const __bf16* WT, int K, int lane, int n0, int k0) {
    int c = lane & 15, half = lane >> 4;
    const __bf16* p = WT + (size_t)(n0 + c) * K + k0 + half * 8;
    union { v16bf v; v8bf h[2]; } u;
    u.h[0] = *(const v8bf*)(p);
    u.h[1] = *(const v8bf*)(p + 16);
    return u.v;
}

// ---------------------------------------------------------------------------
// Weight prep: src fp32 [K][N] row-major -> dst bf16 [N][K] (B-ready layout)
// ---------------------------------------------------------------------------
__global__ void transpose_to_bf16(const float* __restrict__ src, __bf16* __restrict__ dst,
                                  int K, int N) {
    int idx = blockIdx.x * 256 + threadIdx.x;
    if (idx < K * N) {
        int n = idx / K, k = idx % K;
        dst[idx] = (__bf16)src[k * N + n];
    }
}

// ---------------------------------------------------------------------------
// Batched row GEMM: out[16384][256] = A[16384][256] @ W[256][256] (+bias)
// One block = 16-row stripe, 16 waves = 16 column tiles, bf16 WMMA, f32 acc.
// ---------------------------------------------------------------------------
__global__ __launch_bounds__(512) void gemm_rows16(const float* __restrict__ A,
                                                   const __bf16* __restrict__ WT,
                                                   const float* __restrict__ bias,
                                                   float* __restrict__ out) {
    __shared__ __bf16 As[16][256];
    int t = threadIdx.x;
    size_t r0 = (size_t)blockIdx.x * 16;
    const float* ap = A + r0 * 256;
    #pragma unroll
    for (int idx = t; idx < 16 * 256; idx += 512)
        As[idx >> 8][idx & 255] = (__bf16)ap[idx];
    __syncthreads();

    int lane = t & 31, wv = t >> 5;
    int n0 = wv * 16, c = lane & 15, half = lane >> 4;
    v8f acc = {0.f, 0.f, 0.f, 0.f, 0.f, 0.f, 0.f, 0.f};
    #pragma unroll
    for (int kk = 0; kk < 8; kk++) {
        v16bf a = load_a16(&As[0][0], 256, lane, kk * 32);
        v16bf b = load_b16(WT, 256, lane, n0, kk * 32);
        acc = WMMA_BF16(a, b, acc);
    }
    float bv = bias ? bias[n0 + c] : 0.f;
    #pragma unroll
    for (int i = 0; i < 8; i++)
        out[(r0 + i + 8 * half) * 256 + n0 + c] = acc[i] + bv;
}

// ---------------------------------------------------------------------------
// Per-token kernel: k_=k@Ws, v=k_@Ws, pos-MLP, attn-MLP, masked softmax over
// M=16, weighted sum -> x[token][256].  1 block = 1 token, 16 waves.
// ---------------------------------------------------------------------------
__global__ __launch_bounds__(512) void token_kernel(
    const float* __restrict__ k_in, const float* __restrict__ pos,
    const int* __restrict__ mask,
    const float* __restrict__ Wp1, const float* __restrict__ bp1,
    const float* __restrict__ bp2, const float* __restrict__ ba1,
    const float* __restrict__ ba2,
    const __bf16* __restrict__ WsT, const __bf16* __restrict__ Wa1T,
    const __bf16* __restrict__ Wa2T, const __bf16* __restrict__ Wp2T,
    const float* __restrict__ qbuf, float* __restrict__ xbuf) {

    __shared__ __bf16 kbf[16][256];   // k (bf16); reused later for attn input t
    __shared__ __bf16 k2[16][256];    // k_ (bf16)
    __shared__ __bf16 h1[16][32];     // relu(pos@Wp1+bp1)
    __shared__ __bf16 h2[16][32];     // relu(t@Wa1+ba1)
    __shared__ float  qrow[256];
    __shared__ int    smask[16];

    size_t tok = blockIdx.x;
    int t = threadIdx.x, lane = t & 31, wv = t >> 5;
    int n0 = wv * 16, c = lane & 15, half = lane >> 4;

    // ---- stage k (fp32 -> bf16), q_ row, mask, and tiny pos MLP layer 1 ----
    const float* kp = k_in + tok * 16 * 256;
    #pragma unroll
    for (int idx = t; idx < 16 * 256; idx += 512)
        kbf[idx >> 8][idx & 255] = (__bf16)kp[idx];
    if (t < 256) qrow[t] = qbuf[tok * 256 + t];
    if (t < 16)  smask[t] = mask[tok * 16 + t];
    {   // h1[m][j] = relu(bp1[j] + sum_i pos[m][i]*Wp1[i][j]) ; one output/thread
        int m = t >> 5, j = t & 31;
        const float* pp = pos + (tok * 16 + m) * 4;
        float s = bp1[j] + pp[0] * Wp1[j] + pp[1] * Wp1[32 + j]
                         + pp[2] * Wp1[64 + j] + pp[3] * Wp1[96 + j];
        h1[m][j] = (__bf16)fmaxf(s, 0.f);
    }
    __syncthreads();

    // ---- Ws B-fragments: loaded once, reused for BOTH k@Ws and k_@Ws ----
    v16bf bws[8];
    #pragma unroll
    for (int kk = 0; kk < 8; kk++) bws[kk] = load_b16(WsT, 256, lane, n0, kk * 32);

    // ---- k_ = k @ Ws ----
    v8f kacc = {0.f, 0.f, 0.f, 0.f, 0.f, 0.f, 0.f, 0.f};
    #pragma unroll
    for (int kk = 0; kk < 8; kk++) {
        v16bf a = load_a16(&kbf[0][0], 256, lane, kk * 32);
        kacc = WMMA_BF16(a, bws[kk], kacc);
    }
    #pragma unroll
    for (int i = 0; i < 8; i++) k2[i + 8 * half][n0 + c] = (__bf16)kacc[i];

    // ---- posf = h1 @ Wp2 + bp2 (K=32, one WMMA) ----
    v8f pacc = {0.f, 0.f, 0.f, 0.f, 0.f, 0.f, 0.f, 0.f};
    {
        v16bf a = load_a16(&h1[0][0], 32, lane, 0);
        v16bf b = load_b16(Wp2T, 32, lane, n0, 0);
        pacc = WMMA_BF16(a, b, pacc);
        float bv = bp2[n0 + c];
        #pragma unroll
        for (int i = 0; i < 8; i++) pacc[i] += bv;
    }
    __syncthreads();   // k2 complete

    // ---- v = k_ @ Ws (reuse register-resident Ws fragments) ----
    v8f vacc = {0.f, 0.f, 0.f, 0.f, 0.f, 0.f, 0.f, 0.f};
    #pragma unroll
    for (int kk = 0; kk < 8; kk++) {
        v16bf a = load_a16(&k2[0][0], 256, lane, kk * 32);
        vacc = WMMA_BF16(a, bws[kk], vacc);
    }

    // ---- t = k_ - q_ + posf -> bf16 (reuse kbf buffer) ----
    {
        float qv = qrow[n0 + c];
        #pragma unroll
        for (int i = 0; i < 8; i++)
            kbf[i + 8 * half][n0 + c] = (__bf16)(kacc[i] - qv + pacc[i]);
    }
    __syncthreads();   // t complete

    // ---- h = relu(t @ Wa1 + ba1): 16x32 output, waves 0-1 only ----
    if (wv < 2) {
        v8f hacc = {0.f, 0.f, 0.f, 0.f, 0.f, 0.f, 0.f, 0.f};
        #pragma unroll
        for (int kk = 0; kk < 8; kk++) {
            v16bf a = load_a16(&kbf[0][0], 256, lane, kk * 32);
            v16bf b = load_b16(Wa1T, 256, lane, n0, kk * 32);
            hacc = WMMA_BF16(a, b, hacc);
        }
        float bv = ba1[n0 + c];
        #pragma unroll
        for (int i = 0; i < 8; i++)
            h2[i + 8 * half][n0 + c] = (__bf16)fmaxf(hacc[i] + bv, 0.f);
    }
    __syncthreads();

    // ---- attn logits = h @ Wa2 + ba2 (K=32, one WMMA) ----
    v8f aacc = {0.f, 0.f, 0.f, 0.f, 0.f, 0.f, 0.f, 0.f};
    {
        v16bf a = load_a16(&h2[0][0], 32, lane, 0);
        v16bf b = load_b16(Wa2T, 32, lane, n0, 0);
        aacc = WMMA_BF16(a, b, aacc);
    }

    // ---- mask + softmax over M (8 rows in-lane, partner rows via shfl_xor 16)
    float bv2 = ba2[n0 + c];
    float e[8], mx = -3.4e38f;
    #pragma unroll
    for (int i = 0; i < 8; i++) {
        float l = aacc[i] + bv2;
        if (smask[i + 8 * half] == 0) l = -1e9f;
        e[i] = l;
        mx = fmaxf(mx, l);
    }
    mx = fmaxf(mx, __shfl_xor(mx, 16, 32));
    float sum = 0.f;
    #pragma unroll
    for (int i = 0; i < 8; i++) { e[i] = __expf(e[i] - mx); sum += e[i]; }
    sum += __shfl_xor(sum, 16, 32);
    float inv = 1.f / sum;

    // ---- x[c] = sum_m (v + posf)[m][c] * attn[m][c] ----
    float xs = 0.f;
    #pragma unroll
    for (int i = 0; i < 8; i++) xs += (vacc[i] + pacc[i]) * (e[i] * inv);
    xs += __shfl_xor(xs, 16, 32);
    if (half == 0) xbuf[tok * 256 + n0 + c] = xs;
}

// ---------------------------------------------------------------------------
extern "C" void kernel_launch(void* const* d_in, const int* in_sizes, int n_in,
                              void* d_out, int out_size, void* d_ws, size_t ws_size,
                              hipStream_t stream) {
    const float* q    = (const float*)d_in[0];
    const float* k    = (const float*)d_in[1];
    const float* pos  = (const float*)d_in[2];
    const int*   mask = (const int*)d_in[3];
    const float* Ws   = (const float*)d_in[4];
    const float* Wp1  = (const float*)d_in[5];
    const float* bp1  = (const float*)d_in[6];
    const float* Wp2  = (const float*)d_in[7];
    const float* bp2  = (const float*)d_in[8];
    const float* Wa1  = (const float*)d_in[9];
    const float* ba1  = (const float*)d_in[10];
    const float* Wa2  = (const float*)d_in[11];
    const float* ba2  = (const float*)d_in[12];
    const float* Wo   = (const float*)d_in[13];
    const float* bo   = (const float*)d_in[14];

    const int TOKENS = 4 * 4096;   // B*N

    char* wsp = (char*)d_ws;
    __bf16* WsT  = (__bf16*)wsp; wsp += 256 * 256 * 2;
    __bf16* WoT  = (__bf16*)wsp; wsp += 256 * 256 * 2;
    __bf16* Wa1T = (__bf16*)wsp; wsp += 32 * 256 * 2;
    __bf16* Wa2T = (__bf16*)wsp; wsp += 256 * 32 * 2;
    __bf16* Wp2T = (__bf16*)wsp; wsp += 256 * 32 * 2;
    float*  qbuf = (float*)wsp;  wsp += (size_t)TOKENS * 256 * 4;
    float*  xbuf = (float*)wsp;  wsp += (size_t)TOKENS * 256 * 4;

    // Weight prep (bf16, transposed to B-fragment-friendly [N][K])
    transpose_to_bf16<<<(256 * 256 + 255) / 256, 256, 0, stream>>>(Ws, WsT, 256, 256);
    transpose_to_bf16<<<(256 * 256 + 255) / 256, 256, 0, stream>>>(Wo, WoT, 256, 256);
    transpose_to_bf16<<<(256 * 32 + 255) / 256, 256, 0, stream>>>(Wa1, Wa1T, 256, 32);
    transpose_to_bf16<<<(256 * 32 + 255) / 256, 256, 0, stream>>>(Wa2, Wa2T, 32, 256);
    transpose_to_bf16<<<(256 * 32 + 255) / 256, 256, 0, stream>>>(Wp2, Wp2T, 32, 256);

    // q_ = q @ Ws
    gemm_rows16<<<TOKENS / 16, 512, 0, stream>>>(q, WsT, nullptr, qbuf);

    // Per-token attention core
    token_kernel<<<TOKENS, 512, 0, stream>>>(k, pos, mask, Wp1, bp1, bp2, ba1, ba2,
                                             WsT, Wa1T, Wa2T, Wp2T, qbuf, xbuf);

    // out = x @ Wo + bo
    gemm_rows16<<<TOKENS / 16, 512, 0, stream>>>(xbuf, WoT, bo, (float*)d_out);
}
